// Mutual_Attention3D_86818468921914
// MI455X (gfx1250) — matrix operand order repaired and checked by reading, not compiled, not run on hardware
//
#include <hip/hip_runtime.h>
#include <hip/hip_bf16.h>

typedef unsigned int       u32;
typedef unsigned short     u16;
typedef unsigned long long u64;

typedef __attribute__((ext_vector_type(16))) __bf16 v16bf;
typedef __attribute__((ext_vector_type(8)))  float  v8f;
typedef __attribute__((ext_vector_type(2)))  float  v2f;
typedef __attribute__((ext_vector_type(4)))  unsigned int u32x4;
typedef __attribute__((ext_vector_type(4)))  int    i32x4;
typedef __attribute__((ext_vector_type(8)))  int    i32x8;

static constexpr int B_ = 4;
static constexpr int C_ = 512;
static constexpr int N_ = 16384;
static constexpr int H_ = 8;

#if defined(__gfx1250__) && __has_builtin(__builtin_amdgcn_tensor_load_to_lds) && \
    __has_builtin(__builtin_amdgcn_s_wait_tensorcnt)
#define USE_TDM 1
#else
#define USE_TDM 0
#endif

__device__ __forceinline__ u16 f32_to_bf16(float f) {
    u32 u = __float_as_uint(f);
    u32 lsb = (u >> 16) & 1u;
    u += 0x7fffu + lsb;            // round-to-nearest-even
    return (u16)(u >> 16);
}

#if USE_TDM
// Issue one 2-D TDM tile load: global (row-major bf16, row stride `stride_elems`)
// -> LDS at byte address lds_byte, inserting 4 dwords of padding every 32 dwords
// (so a 64-element bf16 row lands on a 36-dword LDS stride).
__device__ __forceinline__ void tdm_load_2d(u32 lds_byte, const u16* gptr,
                                            int tile_d0, int tile_d1, int stride_elems) {
    u64 ga = (u64)(uintptr_t)gptr;
    u32x4 g0;
    g0[0] = 1u;                                        // count=1, user descriptor
    g0[1] = lds_byte;                                  // lds_addr
    g0[2] = (u32)ga;                                   // global_addr[31:0]
    g0[3] = ((u32)(ga >> 32) & 0x01FFFFFFu) | (2u << 30);  // addr[56:32] | type=2
    const u32 TD = 0x40000000u;                        // huge tensor dims: no OOB clipping
    i32x8 g1;
    g1[0] = (int)((1u << 16)      // data_size = 1 -> 2 bytes
                | (1u << 20)      // pad_enable
                | (4u << 22)      // pad_interval: 32 dwords
                | (3u << 25));    // pad_amount: 4 dwords
    g1[1] = (int)((TD & 0xffffu) << 16);                       // tensor_dim0[15:0]
    g1[2] = (int)((TD >> 16) | ((TD & 0xffffu) << 16));        // td0 hi | tensor_dim1 lo
    g1[3] = (int)((TD >> 16) | ((u32)tile_d0 << 16));          // td1 hi | tile_dim0
    g1[4] = (int)tile_d1;                                      // tile_dim1 (tile_dim2=0)
    g1[5] = (int)stride_elems;                                 // tensor_dim0_stride[31:0]
    g1[6] = 0;                                                 // stride hi | dim1_stride
    g1[7] = 0;
    i32x4 z4 = {0, 0, 0, 0};
#if __clang_major__ >= 23
    i32x8 z8 = {0, 0, 0, 0, 0, 0, 0, 0};
    __builtin_amdgcn_tensor_load_to_lds(g0, g1, z4, z4, z8, 0);
#else
    __builtin_amdgcn_tensor_load_to_lds(g0, g1, z4, z4, 0);
#endif
}
#endif

// ---------------------------------------------------------------------------
// Kernel 1: vT[b][n][c] = depthwise_conv3(y)[b][c][n]  (bf16, LDS-tiled
// transpose so both GEMM operands become row-major rectangular tiles)
// ---------------------------------------------------------------------------
__global__ __launch_bounds__(256) void convvT_kernel(const float* __restrict__ y,
                                                     const float* __restrict__ w,
                                                     u16* __restrict__ vT) {
    __shared__ u16 tile[64][68];
    const int n0 = blockIdx.x * 64;
    const int c0 = blockIdx.y * 64;
    const int b  = blockIdx.z;
    const int t  = threadIdx.x;
    {
        const int nl = t & 63, ci = t >> 6;          // coalesced along n
        #pragma unroll 4
        for (int it = 0; it < 16; ++it) {
            const int cl = it * 4 + ci;
            const int c  = c0 + cl;
            const float* yc = y + ((size_t)(b * C_ + c)) * N_;
            const float w0 = w[(2 * C_ + c) * 3 + 0];
            const float w1 = w[(2 * C_ + c) * 3 + 1];
            const float w2 = w[(2 * C_ + c) * 3 + 2];
            const int g = n0 + nl;
            float a  = (g > 0)      ? yc[g - 1] : 0.f;
            float b0 = yc[g];
            float cc = (g < N_ - 1) ? yc[g + 1] : 0.f;
            tile[cl][nl] = f32_to_bf16(w0 * a + w1 * b0 + w2 * cc);
        }
    }
    __syncthreads();
    {
        const int cl = t & 63, ni = t >> 6;          // coalesced along c
        #pragma unroll 4
        for (int it = 0; it < 16; ++it) {
            const int nl = it * 4 + ni;
            vT[((size_t)b * N_ + n0 + nl) * C_ + c0 + cl] = tile[cl][nl];
        }
    }
}

// ---------------------------------------------------------------------------
// Kernel 2: per-(b,h) raw Gram S = Q_raw * K_raw^T (split-K over N) via
// V_WMMA_F32_16X16X4_F32, plus per-channel sum-of-squares (deferred L2 norm).
// ---------------------------------------------------------------------------
#define NT 128
#define QSTR 129

__global__ __launch_bounds__(512) void gram_kernel(const float* __restrict__ x,
                                                   const float* __restrict__ y,
                                                   const float* __restrict__ w,
                                                   float* __restrict__ S,
                                                   float* __restrict__ ssq_q,
                                                   float* __restrict__ ssq_k) {
    __shared__ float Qs[64 * QSTR];
    __shared__ float Ks[64 * QSTR];
    const int tid = threadIdx.x;
    const int bh  = blockIdx.y;
    const int b   = bh >> 3, h = bh & 7;
    const int n0  = blockIdx.x * NT;

    {   // stage Q,K tiles (conv on the fly) + sumsq
        const int row = tid >> 3;
        const int seg = tid & 7;
        const int c   = h * 64 + row;
        const float* xc = x + ((size_t)(b * C_ + c)) * N_;
        const float* yc = y + ((size_t)(b * C_ + c)) * N_;
        const float wq0 = w[c * 3 + 0], wq1 = w[c * 3 + 1], wq2 = w[c * 3 + 2];
        const float wk0 = w[(C_ + c) * 3 + 0], wk1 = w[(C_ + c) * 3 + 1], wk2 = w[(C_ + c) * 3 + 2];
        float sq = 0.f, sk = 0.f;
        const int lc0 = seg * 16;
        int gn = n0 + lc0;
        float xm1 = (gn > 0) ? xc[gn - 1] : 0.f;
        float x0v = xc[gn];
        float ym1 = (gn > 0) ? yc[gn - 1] : 0.f;
        float y0v = yc[gn];
        #pragma unroll 4
        for (int i = 0; i < 16; ++i) {
            int g = gn + i;
            float xp1 = (g < N_ - 1) ? xc[g + 1] : 0.f;
            float yp1 = (g < N_ - 1) ? yc[g + 1] : 0.f;
            float qv = wq0 * xm1 + wq1 * x0v + wq2 * xp1;
            float kv = wk0 * ym1 + wk1 * y0v + wk2 * yp1;
            Qs[row * QSTR + lc0 + i] = qv;
            Ks[row * QSTR + lc0 + i] = kv;
            sq += qv * qv;  sk += kv * kv;
            xm1 = x0v; x0v = xp1; ym1 = y0v; y0v = yp1;
        }
        atomicAdd(&ssq_q[b * C_ + c], sq);
        atomicAdd(&ssq_k[b * C_ + c], sk);
    }
    __syncthreads();

    const int wid = tid >> 5, lane = tid & 31;
    const int wr = wid >> 2, wc = wid & 3;
    const int m  = lane & 15, kh = lane >> 4;
#if __has_builtin(__builtin_amdgcn_wmma_f32_16x16x4_f32)
    v8f acc = {};
    const float* qrow = &Qs[(wr * 16 + m) * QSTR];
    const float* krow = &Ks[(wc * 16 + m) * QSTR];
    #pragma unroll 4
    for (int kk = 0; kk < NT; kk += 4) {
        v2f a, bb;
        a.x  = qrow[kk + kh * 2];
        a.y  = qrow[kk + kh * 2 + 1];
        bb.x = krow[kk + kh * 2];
        bb.y = krow[kk + kh * 2 + 1];
        acc = __builtin_amdgcn_wmma_f32_16x16x4_f32(false, a, false, bb,
                                                    (short)0, acc, false, false);
    }
    float* Sbh = S + (size_t)bh * 64 * 64;
    #pragma unroll
    for (int r = 0; r < 8; ++r) {
        atomicAdd(&Sbh[(wr * 16 + r + kh * 8) * 64 + wc * 16 + m], acc[r]);
    }
#else
    for (int idx = tid; idx < 4096; idx += 512) {
        int cR = idx >> 6, dC = idx & 63;
        float s = 0.f;
        for (int j = 0; j < NT; ++j) s += Qs[cR * QSTR + j] * Ks[dC * QSTR + j];
        atomicAdd(&S[(size_t)bh * 4096 + idx], s);
    }
#endif
}

// ---------------------------------------------------------------------------
// Kernel 3: attn = softmax( S * rsq_q[c] * rsq_k[d] * temperature[h] )
// ---------------------------------------------------------------------------
__global__ __launch_bounds__(64) void softmax_kernel(const float* __restrict__ S,
                                                     const float* __restrict__ ssq_q,
                                                     const float* __restrict__ ssq_k,
                                                     const float* __restrict__ temperature,
                                                     float* __restrict__ attn) {
    const int bh = blockIdx.x;
    const int b = bh >> 3, h = bh & 7;
    const int c = threadIdx.x;
    __shared__ float rk[64];
    float rq = 1.f / fmaxf(sqrtf(ssq_q[b * C_ + h * 64 + c]), 1e-12f);
    rk[c]    = 1.f / fmaxf(sqrtf(ssq_k[b * C_ + h * 64 + c]), 1e-12f);
    __syncthreads();
    const float tmp = temperature[h];
    const float* Srow = S    + ((size_t)bh * 64 + c) * 64;
    float*       Arow = attn + ((size_t)bh * 64 + c) * 64;
    float mx = -3.4e38f;
    for (int d = 0; d < 64; ++d) mx = fmaxf(mx, Srow[d] * rq * rk[d] * tmp);
    float sum = 0.f;
    for (int d = 0; d < 64; ++d) {
        float e = __expf(Srow[d] * rq * rk[d] * tmp - mx);
        Arow[d] = e;  sum += e;
    }
    float inv = 1.f / sum;
    for (int d = 0; d < 64; ++d) Arow[d] *= inv;
}

// ---------------------------------------------------------------------------
// Kernel 4: M_b[o, h*64+d] = sum_c Wproj[o, h*64+c] * attn[b,h,c,d]  (bf16)
// ---------------------------------------------------------------------------
__global__ __launch_bounds__(512) void foldproj_kernel(const float* __restrict__ wproj,
                                                       const float* __restrict__ attn,
                                                       u16* __restrict__ Mb) {
    const int o = blockIdx.x, b = blockIdx.y;
    const int t = threadIdx.x;
    const int h = t >> 6, d = t & 63;
    const float* wp = wproj + (size_t)o * C_ + h * 64;
    const float* a  = attn + ((size_t)(b * H_ + h)) * 64 * 64 + d;
    float s = 0.f;
    #pragma unroll 8
    for (int c = 0; c < 64; ++c) s += wp[c] * a[c * 64];
    Mb[((size_t)b * C_ + o) * C_ + h * 64 + d] = f32_to_bf16(s);
}

// ---------------------------------------------------------------------------
// Kernel 5: out[b] = M_b (512x512 bf16) @ vT_b^T (512x16384 bf16), f32 acc.
// 64(o) x 128(n) block tile, K-step 64, double-buffered LDS fed by the
// Tensor Data Mover (2 descriptors/step, s_wait_tensorcnt), 16 waves each
// computing a 16x32 tile with v_wmma_f32_16x16x32_bf16.
// ---------------------------------------------------------------------------
union FragBF { v16bf v; uint4 q[2]; };

// LDS layout (dwords): A bufs [2][64*36] then B bufs [2][128*36]; a 64-elem
// bf16 row (32 dwords) + 4 dwords TDM padding = 36-dword stride (conflict-free).
#define A_BUF_DW (64 * 36)
#define B_BUF_DW (128 * 36)

__global__ __launch_bounds__(512) void gemm_kernel(const u16* __restrict__ Mb,
                                                   const u16* __restrict__ vT,
                                                   float* __restrict__ out) {
    __shared__ u32 lds[2 * A_BUF_DW + 2 * B_BUF_DW];
    const int tid = threadIdx.x;
    const int b  = blockIdx.z;
    const int o0 = blockIdx.y * 64;
    const int n0 = blockIdx.x * 128;
    const int wid = tid >> 5, lane = tid & 31;
    const int wo = wid >> 2, wn = wid & 3;
    const int m  = lane & 15, kh = lane >> 4;
    const u16* Abase = Mb + ((size_t)b * C_ + o0) * C_;   // row stride 512
    const u16* Bbase = vT + ((size_t)b * N_ + n0) * C_;   // row stride 512
    v8f acc0 = {}, acc1 = {};
    constexpr int KSTEP = 64, NSTEP = C_ / KSTEP;

#if USE_TDM
    const u32 lds_base = (u32)(uintptr_t)(&lds[0]);
    if (wid == 0) {                       // prologue: stage buffer 0
        tdm_load_2d(lds_base, Abase, KSTEP, 64, C_);
        tdm_load_2d(lds_base + 2 * A_BUF_DW * 4, Bbase, KSTEP, 128, C_);
    }
#endif
    for (int s = 0; s < NSTEP; ++s) {
        const int buf = s & 1;
#if USE_TDM
        if (wid == 0) {
            if (s + 1 < NSTEP) {          // issue next tiles, then retire current
                const int nb = (s + 1) & 1;
                tdm_load_2d(lds_base + nb * A_BUF_DW * 4,
                            Abase + (s + 1) * KSTEP, KSTEP, 64, C_);
                tdm_load_2d(lds_base + (2 * A_BUF_DW + nb * B_BUF_DW) * 4,
                            Bbase + (s + 1) * KSTEP, KSTEP, 128, C_);
                __builtin_amdgcn_s_wait_tensorcnt(2);
            } else {
                __builtin_amdgcn_s_wait_tensorcnt(0);
            }
        }
        __syncthreads();
#else
        __syncthreads();
        {   // synchronous fallback staging (coalesced uint4 copies)
            const u32* Au = (const u32*)Abase;
            const u32* Bu = (const u32*)Bbase;
            int i = tid;                  // A: 64 rows x 8 uint4
            {
                int row = i >> 3, q = i & 7;
                *(uint4*)&lds[buf * A_BUF_DW + row * 36 + q * 4] =
                    *(const uint4*)&Au[row * (C_ / 2) + s * (KSTEP / 2) + q * 4];
            }
            #pragma unroll
            for (int it = 0; it < 2; ++it, i += 512) {  // B: 128 rows x 8 uint4
                int row = i >> 3, q = i & 7;
                *(uint4*)&lds[2 * A_BUF_DW + buf * B_BUF_DW + row * 36 + q * 4] =
                    *(const uint4*)&Bu[row * (C_ / 2) + s * (KSTEP / 2) + q * 4];
            }
        }
        __syncthreads();
#endif
        const u32* Ab = &lds[buf * A_BUF_DW];
        const u32* Bb = &lds[2 * A_BUF_DW + buf * B_BUF_DW];
        #pragma unroll
        for (int ks = 0; ks < 2; ++ks) {
            FragBF a, b0, b1;
            // A 16x32 bf16: lanes0-15 K=0-15; lanes16-31 K=16-31
            const u32* ap = Ab + (wo * 16 + m) * 36 + ks * 16 + kh * 8;
            a.q[0] = *(const uint4*)ap;
            a.q[1] = *(const uint4*)(ap + 4);
            // B 32x16 bf16: lanes0-15 K=0-15; lanes16-31 K=16-31
            const u32* bp0 = Bb + (wn * 32 + m) * 36 + ks * 16 + kh * 8;
            b0.q[0] = *(const uint4*)bp0;
            b0.q[1] = *(const uint4*)(bp0 + 4);
            const u32* bp1 = Bb + (wn * 32 + 16 + m) * 36 + ks * 16 + kh * 8;
            b1.q[0] = *(const uint4*)bp1;
            b1.q[1] = *(const uint4*)(bp1 + 4);
            acc0 = __builtin_amdgcn_wmma_f32_16x16x32_bf16(false, a.v, false, b0.v,
                                                           (short)0, acc0, false, false);
            acc1 = __builtin_amdgcn_wmma_f32_16x16x32_bf16(false, a.v, false, b1.v,
                                                           (short)0, acc1, false, false);
        }
        __syncthreads();
    }
    float* outb = out + ((size_t)b * C_ + o0 + wo * 16) * N_ + n0 + wn * 32;
    #pragma unroll
    for (int r = 0; r < 8; ++r) {
        int row = r + kh * 8;
        outb[(size_t)row * N_ + m]      = acc0[r];
        outb[(size_t)row * N_ + 16 + m] = acc1[r];
    }
}

// ---------------------------------------------------------------------------
// Launcher
// ---------------------------------------------------------------------------
extern "C" void kernel_launch(void* const* d_in, const int* in_sizes, int n_in,
                              void* d_out, int out_size, void* d_ws, size_t ws_size,
                              hipStream_t stream) {
    const float* x     = (const float*)d_in[0];
    const float* y     = (const float*)d_in[1];
    const float* w_qkv = (const float*)d_in[2];
    const float* temp  = (const float*)d_in[3];
    const float* wproj = (const float*)d_in[4];
    float* out = (float*)d_out;

    char* ws = (char*)d_ws;
    const size_t vbytes = (size_t)B_ * C_ * N_ * sizeof(u16);        // 64 MiB
    const size_t sbytes = (size_t)B_ * H_ * 64 * 64 * sizeof(float); // 512 KiB
    const size_t qbytes = (size_t)B_ * C_ * sizeof(float);           // 8 KiB
    u16*   vT    = (u16*)ws;
    float* S     = (float*)(ws + vbytes);
    float* ssq_q = (float*)(ws + vbytes + sbytes);
    float* ssq_k = (float*)(ws + vbytes + sbytes + qbytes);
    float* attn  = (float*)(ws + vbytes + sbytes + 2 * qbytes);
    u16*   Mb    = (u16*)(ws + vbytes + 2 * sbytes + 2 * qbytes);

    hipMemsetAsync(S, 0, sbytes + 2 * qbytes, stream);

    convvT_kernel<<<dim3(N_ / 64, C_ / 64, B_), 256, 0, stream>>>(y, w_qkv, vT);
    gram_kernel<<<dim3(N_ / NT, B_ * H_), 512, 0, stream>>>(x, y, w_qkv, S, ssq_q, ssq_k);
    softmax_kernel<<<dim3(B_ * H_), 64, 0, stream>>>(S, ssq_q, ssq_k, temp, attn);
    foldproj_kernel<<<dim3(C_, B_), 512, 0, stream>>>(wproj, attn, Mb);
    gemm_kernel<<<dim3(N_ / 128, C_ / 64, B_), 512, 0, stream>>>(Mb, vT, out);
}
